// KPConv_89515708383489
// MI455X (gfx1250) — matrix operand order repaired and checked by reading, not compiled, not run on hardware
//
#include <hip/hip_runtime.h>

// ---------------------------------------------------------------------------
// KPConv on MI455X (gfx1250, wave32): bf16 WMMA (v_wmma_f32_16x16x32_bf16)
// Block = 256 threads (8 waves) handles 16 query points.
//   Phase A: gather neighbor feats -> LDS bf16 [n][c][h]; weights -> [n][k][h]
//   Phase B: per-query 16x16x32 WMMA: weighted[n](16x64) = w(16x32) @ F(32x64)
//   Phase C: batched 16x960x128 GEMM vs pre-packed bf16 wts (30 K-steps/wave)
// ---------------------------------------------------------------------------

typedef __attribute__((ext_vector_type(16))) __bf16 v16bf;
typedef __attribute__((ext_vector_type(8)))  __bf16 v8bf;
typedef __attribute__((ext_vector_type(8)))  float  v8f;

constexpr int H    = 32;     // neighbors
constexpr int KP   = 15;     // kernel points
constexpr int CIN  = 64;
constexpr int COUT = 128;
constexpr int QT   = 16;     // queries per block
constexpr int KD   = KP * CIN;      // 960 flattened contraction dim
constexpr int KSTEPS = KD / 32;     // 30
constexpr int NTILES = COUT / 16;   // 8

#define KP_EXTENT_INV (1.0f / 1.2f)

// LDS layout (bytes)
constexpr int OFF_SF  = 0;                      // bf16 [16][64][32]  = 65536
constexpr int OFF_SW  = OFF_SF  + 16*CIN*H*2;   // bf16 [16][16][32]  = 16384
constexpr int OFF_SA  = OFF_SW  + 16*16*H*2;    // bf16 [16][960]     = 30720
constexpr int OFF_NBR = OFF_SA  + 16*KD*2;      // f32  [16][32][3]   = 6144
constexpr int OFF_KP  = OFF_NBR + 16*H*3*4;     // f32  48 (pad 192)
constexpr int OFF_QP  = OFF_KP  + 192;          // f32  48 (pad 192)
constexpr int OFF_IDX = OFF_QP  + 192;          // i32  512           = 2048
constexpr int SMEM_BYTES = OFF_IDX + QT*H*4;    // 121216 B  (< 320 KB WGP LDS)

__device__ __forceinline__ unsigned pk2(float lo, float hi) {
  union { __bf16 b[2]; unsigned u; } x;
  x.b[0] = (__bf16)lo; x.b[1] = (__bf16)hi;
  return x.u;
}

// ---------------------------------------------------------------------------
// Prep: wts f32 [15][64][128] -> bf16 in d_ws, pre-swizzled into per-lane
// contiguous B-operand tiles: index = (((s*8 + j)*32 + lane)*16 + e)
// where lane<16 : N = j*16+lane,    K = s*32 + e
//       lane>=16: N = j*16+lane-16, K = s*32 + 16 + e   (contiguous K per lane half; A uses same order)
// ---------------------------------------------------------------------------
__global__ __launch_bounds__(256) void kpconv_prep_wts(
    const float* __restrict__ wts, __bf16* __restrict__ wbf) {
  int t = blockIdx.x * 256 + threadIdx.x;
  if (t >= KSTEPS * NTILES * 32 * 16) return;
  int e    = t & 15;
  int lane = (t >> 4) & 31;
  int j    = (t >> 9) & 7;
  int s    = t >> 12;
  int klocal = ((lane >= 16) ? 16 : 0) + e;
  int kk = s * 32 + klocal;          // 0..959
  int n  = j * 16 + (lane & 15);     // 0..127
  int kp = kk >> 6, c = kk & 63;
  wbf[t] = (__bf16)wts[(kp * CIN + c) * COUT + n];
}

__global__ __launch_bounds__(256) void kpconv_wmma_kernel(
    const float* __restrict__ qpts, const float* __restrict__ spts,
    const int*   __restrict__ nidx, const float* __restrict__ feats,
    const float* __restrict__ kpts, const __bf16* __restrict__ wbf,
    float* __restrict__ out, int N, int M) {
  extern __shared__ char smem[];
  __bf16* sF   = (__bf16*)(smem + OFF_SF);   // [n][c][h]
  __bf16* sW   = (__bf16*)(smem + OFF_SW);   // [n][k][h]  (k padded to 16)
  __bf16* sA   = (__bf16*)(smem + OFF_SA);   // [n][kk=960]
  float*  sNbr = (float*) (smem + OFF_NBR);  // [n][h][3]
  float*  sKP  = (float*) (smem + OFF_KP);
  float*  sQP  = (float*) (smem + OFF_QP);
  int*    sIdx = (int*)   (smem + OFF_IDX);

  const int tid  = threadIdx.x;
  const int wave = tid >> 5;
  const int lane = tid & 31;
  const int qbase = blockIdx.x * QT;
  int nq = N - qbase; if (nq > QT) nq = QT;

  // ---- Phase 0: stage small arrays ---------------------------------------
  for (int i = tid; i < KP * 3; i += 256) sKP[i] = kpts[i];
  for (int i = tid; i < QT * 3; i += 256)
    sQP[i] = (i < nq * 3) ? qpts[(size_t)qbase * 3 + i] : 0.0f;
  for (int i = tid; i < QT * H; i += 256)
    sIdx[i] = (i < nq * H) ? nidx[(size_t)qbase * H + i] : 0x7fffffff;
  __syncthreads();

  // ---- Phase 1: gather features (bf16) + neighbor coords -----------------
  {
    const int n  = tid >> 4;
    const int hp = tid & 15;
    const int h0 = hp * 2, h1 = h0 + 1;
    const int i0 = sIdx[n * H + h0], i1 = sIdx[n * H + h1];
    const bool v0 = (unsigned)i0 < (unsigned)M;
    const bool v1 = (unsigned)i1 < (unsigned)M;
    const size_t o0 = (size_t)(v0 ? i0 : 0);
    const size_t o1 = (size_t)(v1 ? i1 : 0);
    const float qx = sQP[n*3+0], qy = sQP[n*3+1], qz = sQP[n*3+2];
    const float* s0 = spts + o0 * 3;
    const float* s1 = spts + o1 * 3;
    const float big = 1.0e6f;  // shadow point: weight -> 0
    sNbr[(n*H+h0)*3+0] = v0 ? s0[0]-qx : big;
    sNbr[(n*H+h0)*3+1] = v0 ? s0[1]-qy : big;
    sNbr[(n*H+h0)*3+2] = v0 ? s0[2]-qz : big;
    sNbr[(n*H+h1)*3+0] = v1 ? s1[0]-qx : big;
    sNbr[(n*H+h1)*3+1] = v1 ? s1[1]-qy : big;
    sNbr[(n*H+h1)*3+2] = v1 ? s1[2]-qz : big;
    const float4* f0 = (const float4*)(feats + o0 * CIN);
    const float4* f1 = (const float4*)(feats + o1 * CIN);
    const float4 z4 = make_float4(0.f, 0.f, 0.f, 0.f);
#pragma unroll
    for (int c4 = 0; c4 < CIN / 4; ++c4) {
      float4 a = v0 ? f0[c4] : z4;
      float4 b = v1 ? f1[c4] : z4;
      unsigned* dst = (unsigned*)&sF[n * (CIN * H) + (c4 * 4) * H + h0];
      dst[0]        = pk2(a.x, b.x);   // c+0, pair (h0,h1)
      dst[H / 2]    = pk2(a.y, b.y);   // c+1
      dst[H]        = pk2(a.z, b.z);   // c+2
      dst[3 * H/2]  = pk2(a.w, b.w);   // c+3
    }
  }
  __syncthreads();

  // ---- Phase 2: influence weights w[n][k][h] (bf16, k padded to 16) ------
  {
    const int n = tid >> 4;
    const int k = tid & 15;
    const bool kv = (k < KP);
    float kx = 0.f, ky = 0.f, kz = 0.f;
    if (kv) { kx = sKP[k*3]; ky = sKP[k*3+1]; kz = sKP[k*3+2]; }
#pragma unroll
    for (int hp = 0; hp < H / 2; ++hp) {
      float w0 = 0.f, w1 = 0.f;
      if (kv) {
        const float* p = &sNbr[(n * H + hp * 2) * 3];
        float dx = p[0]-kx, dy = p[1]-ky, dz = p[2]-kz;
        w0 = fmaxf(1.f - sqrtf(dx*dx + dy*dy + dz*dz) * KP_EXTENT_INV, 0.f);
        dx = p[3]-kx; dy = p[4]-ky; dz = p[5]-kz;
        w1 = fmaxf(1.f - sqrtf(dx*dx + dy*dy + dz*dz) * KP_EXTENT_INV, 0.f);
      }
      *(unsigned*)&sW[n * (16 * H) + k * H + hp * 2] = pk2(w0, w1);
    }
  }
  __syncthreads();

  // ---- Phase 3: step C — weighted[n](16x64) = w(16x32) @ F(32x64) --------
  // A lane layout (16-bit 16x32): lane<16 -> K 0..15; lane>=16 -> K 16..31 (matches B)
  {
    const int row  = lane & 15;               // A row = kernel point
    const int hsel = (lane < 16) ? 0 : 16;
    const int csel = lane & 15;
    const int hs2  = (lane < 16) ? 0 : 16;    // B lane half: K 0..15 / 16..31
    const int kb   = (lane < 16) ? 0 : 8;     // D row base
#pragma unroll
    for (int qq = 0; qq < 2; ++qq) {
      const int n = wave * 2 + qq;
      const __bf16* ap = &sW[n * (16 * H) + row * H + hsel];
      v8bf alo = *(const v8bf*)ap;
      v8bf ahi = *(const v8bf*)(ap + 8);
      v16bf A;
#pragma unroll
      for (int e = 0; e < 8; ++e) { A[e] = alo[e]; A[e + 8] = ahi[e]; }
#pragma unroll
      for (int ct = 0; ct < 4; ++ct) {
        const int c = ct * 16 + csel;
        v16bf B = *(const v16bf*)&sF[n * (CIN * H) + c * H + hs2];
        v8f acc = {};
        acc = __builtin_amdgcn_wmma_f32_16x16x32_bf16(
            false, A, false, B, (short)0, acc, false, false);
#pragma unroll
        for (int v = 0; v < 8; ++v) {
          const int k = kb + v;
          if (k < KP) sA[n * KD + k * CIN + c] = (__bf16)acc[v];
        }
      }
    }
  }
  __syncthreads();

  // ---- Phase 4: step D — out(16x128) = sA(16x960) @ Wflat(960x128) -------
  // Each wave owns one 16-wide Cout tile; 30 K-steps of 16x16x32 WMMA.
  {
    const int j    = wave;
    const int qrow = lane & 15;
    const int ksel = (lane < 16) ? 0 : 16;
    const __bf16* bp = wbf + ((size_t)j * 32 + lane) * 16;
    constexpr size_t BSTRIDE = (size_t)NTILES * 32 * 16;  // per K-step
    v8f acc = {};
#pragma unroll
    for (int s = 0; s < KSTEPS; ++s) {
      const __bf16* ap = &sA[qrow * KD + s * 32 + ksel];
      v8bf alo = *(const v8bf*)ap;
      v8bf ahi = *(const v8bf*)(ap + 8);
      v16bf A;
#pragma unroll
      for (int e = 0; e < 8; ++e) { A[e] = alo[e]; A[e + 8] = ahi[e]; }
      v16bf B = *(const v16bf*)(bp + (size_t)s * BSTRIDE);
      if (s + 1 < KSTEPS)
        __builtin_prefetch(bp + (size_t)(s + 1) * BSTRIDE, 0, 0);
      acc = __builtin_amdgcn_wmma_f32_16x16x32_bf16(
          false, A, false, B, (short)0, acc, false, false);
    }
    const int ocol = j * 16 + (lane & 15);
    const int qb2  = (lane < 16) ? 0 : 8;
#pragma unroll
    for (int v = 0; v < 8; ++v) {
      const int q = qb2 + v;
      if (q < nq) out[(size_t)(qbase + q) * COUT + ocol] = acc[v];
    }
  }
}

extern "C" void kernel_launch(void* const* d_in, const int* in_sizes, int n_in,
                              void* d_out, int out_size, void* d_ws, size_t ws_size,
                              hipStream_t stream) {
  const float* qpts  = (const float*)d_in[0];   // [N,3]
  const float* spts  = (const float*)d_in[1];   // [M,3]
  const int*   nidx  = (const int*)  d_in[2];   // [N,32]
  const float* feats = (const float*)d_in[3];   // [M,64]
  const float* kpts  = (const float*)d_in[4];   // [15,3]
  const float* wts   = (const float*)d_in[5];   // [15,64,128]
  float* out = (float*)d_out;

  const int N = in_sizes[0] / 3;
  const int M = in_sizes[3] / CIN;

  __bf16* wbf = (__bf16*)d_ws;  // 245,760 B of pre-swizzled bf16 weights

  const int prep_elems = KSTEPS * NTILES * 32 * 16;  // 122880
  kpconv_prep_wts<<<(prep_elems + 255) / 256, 256, 0, stream>>>(wts, wbf);

  const int blocks = (N + QT - 1) / QT;  // 6250
  kpconv_wmma_kernel<<<blocks, 256, SMEM_BYTES, stream>>>(
      qpts, spts, nidx, feats, kpts, wbf, out, N, M);
}
